// GRULayer_12292196401855
// MI455X (gfx1250) — matrix operand order repaired and checked by reading, not compiled, not run on hardware
//
#include <hip/hip_runtime.h>
#include <hip/hip_bf16.h>

// Problem sizes (fixed by the reference).
#define Bb 256
#define Tt 512
#define Ff 128
#define Hh 256
#define H3 768
#define NBLK 32

typedef __attribute__((ext_vector_type(16))) __bf16 v16bf;
typedef __attribute__((ext_vector_type(8)))  __bf16 v8bf;
typedef __attribute__((ext_vector_type(4)))  __bf16 v4bf;
typedef __attribute__((ext_vector_type(8)))  float  v8f;

union FragU { v16bf v; v8bf h[2]; };

// A-matrix fragment (16x32 bf16): lane chunk1 at K0+8*half, chunk2 at +16.
__device__ __forceinline__ v16bf load_frag_a(const __bf16* __restrict__ p) {
  FragU f;
  f.h[0] = *(const v8bf*)(p);
  f.h[1] = *(const v8bf*)(p + 16);
  return f.v;
}
// B-matrix fragment (32x16 bf16): lane chunk1 at K0+8*half, chunk2 at +16.
__device__ __forceinline__ v16bf load_frag_b(const __bf16* __restrict__ p) {
  FragU f;
  f.h[0] = *(const v8bf*)(p);
  f.h[1] = *(const v8bf*)(p + 16);
  return f.v;
}

__device__ __forceinline__ float sigm(float x) {
  return 1.0f / (1.0f + __expf(-x));
}
// Overflow-safe tanh.
__device__ __forceinline__ float tanh_f(float x) {
  return 1.0f - 2.0f / (__expf(2.0f * x) + 1.0f);
}

// ---------------- prep kernels ----------------

// x (B,T,F) f32  ->  xb (T,B,F) bf16
__global__ void conv_x_kernel(const float* __restrict__ x, __bf16* __restrict__ xb) {
  const int F4 = Ff / 4;
  int id = blockIdx.x * blockDim.x + threadIdx.x;   // over T*B*F/4
  if (id >= Tt * Bb * F4) return;
  int f4   = id % F4;
  int rest = id / F4;
  int b    = rest % Bb;
  int t    = rest / Bb;
  const float4 v = *(const float4*)(x + ((size_t)b * Tt + t) * Ff + 4 * f4);
  v4bf o;
  o[0] = (__bf16)v.x; o[1] = (__bf16)v.y; o[2] = (__bf16)v.z; o[3] = (__bf16)v.w;
  *(v4bf*)(xb + (size_t)id * 4) = o;
}

// src (K,768) f32  ->  dst (768,K) bf16   (N-major so B fragments are contiguous)
__global__ void conv_w_kernel(const float* __restrict__ src, __bf16* __restrict__ dst,
                              int K, int total) {
  int id = blockIdx.x * blockDim.x + threadIdx.x;
  if (id >= total) return;
  int k = id % K;
  int n = id / K;
  dst[id] = (__bf16)src[(size_t)k * H3 + n];
}

// Zero h0 (bf16) and the grid-barrier counter (re-run every call: deterministic).
__global__ void init_state_kernel(__bf16* __restrict__ hb, unsigned* __restrict__ cnt) {
  int id = blockIdx.x * blockDim.x + threadIdx.x;
  if (id < Bb * Hh) hb[id] = (__bf16)0.0f;
  if (id == 0) *cnt = 0u;
}

// ---------------- persistent GRU kernel ----------------
// 32 blocks x 256 threads (8 waves). Two blocks share one 16-wide column strip
// (j0 = (bid>>1)*16); each block covers half of the batch rows. Each wave owns
// one 16x16 tile of h for ALL timesteps:
//   - its 36 loop-invariant weight B-fragments live in VGPRs (288 regs)
//   - its 8 f32 h elements live in VGPRs
// so the steady-state loop is A-loads + 36 WMMAs + gate math + grid barrier.
__global__ __launch_bounds__(256) void gru_persist(
    const __bf16* __restrict__ xb,    // (T,B,F) bf16
    const __bf16* __restrict__ Wt,    // (3H,F) bf16 (N-major)
    const __bf16* __restrict__ Rt,    // (3H,H) bf16 (N-major)
    const float*  __restrict__ bias,  // (2,768) f32
    __bf16* __restrict__ hb0,         // (B,H) bf16 ping (zeroed)
    __bf16* __restrict__ hb1,         // (B,H) bf16 pong
    float*  __restrict__ out,         // (T,B,H) f32
    unsigned* __restrict__ cnt)       // grid barrier counter (zeroed)
{
  const int bid   = blockIdx.x;
  const int j0    = (bid >> 1) << 4;          // column block in H
  const int ihalf = bid & 1;                  // which half of the batch
  const int wave  = threadIdx.x >> 5;
  const int lane  = threadIdx.x & 31;
  const int i0    = ihalf * 128 + wave * 16;  // row block in B
  const int half  = lane >> 4;
  const int nl    = lane & 15;

  // ---- hoist ALL weight B-fragments into registers (loop-invariant) ----
  v16bf bw[3][4];   // x-path weights: gate g, K-chunk kk
  v16bf br[3][8];   // h-path weights: gate g, K-chunk kk
#pragma unroll
  for (int g = 0; g < 3; ++g) {
    const __bf16* wrow = Wt + (size_t)(g * Hh + j0 + nl) * Ff + 8 * half;
#pragma unroll
    for (int kk = 0; kk < 4; ++kk)
      bw[g][kk] = load_frag_b(wrow + kk * 32);
    const __bf16* rrow = Rt + (size_t)(g * Hh + j0 + nl) * Hh + 8 * half;
#pragma unroll
    for (int kk = 0; kk < 8; ++kk)
      br[g][kk] = load_frag_b(rrow + kk * 32);
  }

  // Per-lane constants for the gate epilogue.
  const int   n   = j0 + nl;
  const float ibz = bias[n],      ibr = bias[Hh + n],      ibh = bias[2 * Hh + n];
  const float rbz = bias[H3 + n], rbr = bias[H3 + Hh + n], rbh = bias[H3 + 2 * Hh + n];

  // h elements owned by this lane (C-layout rows m = i0 + e + 8*half, col n).
  float hp[8];
#pragma unroll
  for (int e = 0; e < 8; ++e) hp[e] = 0.0f;

  for (int t = 0; t < Tt; ++t) {
    const __bf16* xbt   = xb + (size_t)t * Bb * Ff;
    const __bf16* hprev = (t & 1) ? hb1 : hb0;
    __bf16*       hnext = (t & 1) ? hb0 : hb1;

    // Prefetch next step's x rows while we compute this one.
    if (t + 1 < Tt)
      __builtin_prefetch(xb + (size_t)(t + 1) * Bb * Ff + (size_t)(i0 + nl) * Ff + 64 * half, 0, 0);

    v8f ax0 = {}, ax1 = {}, ax2 = {};
    v8f ah0 = {}, ah1 = {}, ah2 = {};

    // ---- mx: K = F = 128, 4 chunks; B from registers ----
    {
      const __bf16* arow = xbt + (size_t)(i0 + nl) * Ff + 8 * half;
#pragma unroll
      for (int kk = 0; kk < 4; ++kk) {
        v16bf a = load_frag_a(arow + kk * 32);
        ax0 = __builtin_amdgcn_wmma_f32_16x16x32_bf16(false, a, false, bw[0][kk], (short)0, ax0, false, false);
        ax1 = __builtin_amdgcn_wmma_f32_16x16x32_bf16(false, a, false, bw[1][kk], (short)0, ax1, false, false);
        ax2 = __builtin_amdgcn_wmma_f32_16x16x32_bf16(false, a, false, bw[2][kk], (short)0, ax2, false, false);
      }
    }
    // ---- mh: K = H = 256, 8 chunks; B from registers ----
    {
      const __bf16* hrow = hprev + (size_t)(i0 + nl) * Hh + 8 * half;
#pragma unroll
      for (int kk = 0; kk < 8; ++kk) {
        v16bf a = load_frag_a(hrow + kk * 32);
        ah0 = __builtin_amdgcn_wmma_f32_16x16x32_bf16(false, a, false, br[0][kk], (short)0, ah0, false, false);
        ah1 = __builtin_amdgcn_wmma_f32_16x16x32_bf16(false, a, false, br[1][kk], (short)0, ah1, false, false);
        ah2 = __builtin_amdgcn_wmma_f32_16x16x32_bf16(false, a, false, br[2][kk], (short)0, ah2, false, false);
      }
    }

    // ---- gates; h carried in registers; states written nontemporal ----
    float* out_t = out + (size_t)t * Bb * Hh;
#pragma unroll
    for (int e = 0; e < 8; ++e) {
      const int m = i0 + e + 8 * half;
      const size_t idx = (size_t)m * Hh + n;
      const float z  = sigm(ax0[e] + ah0[e] + ibz + rbz);
      const float r  = sigm(ax1[e] + ah1[e] + ibr + rbr);
      const float c  = tanh_f(ax2[e] + ibh + r * (ah2[e] + rbh));
      const float hn = z * hp[e] + (1.0f - z) * c;
      hp[e] = hn;
      __builtin_nontemporal_store(hn, out_t + idx);
      hnext[idx] = (__bf16)hn;
    }

    // ---- device-scope grid barrier between timesteps ----
    __threadfence();          // make this block's h stores visible agent-wide
    __syncthreads();
    if (threadIdx.x == 0) {
      __hip_atomic_fetch_add(cnt, 1u, __ATOMIC_RELAXED, __HIP_MEMORY_SCOPE_AGENT);
      const unsigned target = (unsigned)NBLK * (unsigned)(t + 1);
      while (__hip_atomic_load(cnt, __ATOMIC_RELAXED, __HIP_MEMORY_SCOPE_AGENT) < target)
        __builtin_amdgcn_s_sleep(2);
    }
    __syncthreads();
    __threadfence();          // acquire side: invalidate before reading new h
  }
}

// ---------------- host launcher ----------------
extern "C" void kernel_launch(void* const* d_in, const int* in_sizes, int n_in,
                              void* d_out, int out_size, void* d_ws, size_t ws_size,
                              hipStream_t stream) {
  const float* x     = (const float*)d_in[0];   // (B,T,F)
  const float* kern  = (const float*)d_in[1];   // (F,3H)
  const float* rkern = (const float*)d_in[2];   // (H,3H)
  const float* bias  = (const float*)d_in[3];   // (2,3H)
  float* out = (float*)d_out;                   // (T,B,H)

  // Carve workspace (~34.4 MB needed).
  char* p = (char*)d_ws;
  auto carve = [&](size_t bytes) -> char* {
    char* r = p;
    p += (bytes + 255) & ~(size_t)255;
    return r;
  };
  __bf16* xb  = (__bf16*)carve((size_t)Tt * Bb * Ff * sizeof(__bf16));  // (T,B,F)
  __bf16* Wt  = (__bf16*)carve((size_t)H3 * Ff * sizeof(__bf16));       // (3H,F)
  __bf16* Rt  = (__bf16*)carve((size_t)H3 * Hh * sizeof(__bf16));       // (3H,H)
  __bf16* hb0 = (__bf16*)carve((size_t)Bb * Hh * sizeof(__bf16));       // h ping
  __bf16* hb1 = (__bf16*)carve((size_t)Bb * Hh * sizeof(__bf16));       // h pong
  unsigned* cnt = (unsigned*)carve(256);                                 // barrier

  // Prep: convert/transpose + zero state (re-run each call: deterministic).
  {
    int nthr = Tt * Bb * (Ff / 4);
    conv_x_kernel<<<(nthr + 255) / 256, 256, 0, stream>>>(x, xb);
  }
  {
    int total = H3 * Ff;
    conv_w_kernel<<<(total + 255) / 256, 256, 0, stream>>>(kern, Wt, Ff, total);
  }
  {
    int total = H3 * Hh;
    conv_w_kernel<<<(total + 255) / 256, 256, 0, stream>>>(rkern, Rt, Hh, total);
  }
  {
    int total = Bb * Hh;
    init_state_kernel<<<(total + 255) / 256, 256, 0, stream>>>(hb0, cnt);
  }

  // One persistent kernel runs the whole recurrence (32 blocks are guaranteed
  // co-resident on MI455X, so the atomic grid barrier is safe).
  gru_persist<<<NBLK, 256, 0, stream>>>(xb, Wt, Rt, bias, hb0, hb1, out, cnt);
}
